// EncoderLayer_24936580120808
// MI455X (gfx1250) — matrix operand order repaired and checked by reading, not compiled, not run on hardware
//
#include <hip/hip_runtime.h>

// ---------------------------------------------------------------------------
// Types for CDNA5 WMMA (wave32, 16x16x32 bf16 -> f32 accumulate)
// ---------------------------------------------------------------------------
typedef __bf16 bf16_t;
typedef __attribute__((ext_vector_type(16))) __bf16 v16bf;
typedef __attribute__((ext_vector_type(8)))  __bf16 v8bf;
typedef __attribute__((ext_vector_type(4)))  __bf16 v4bf;
typedef __attribute__((ext_vector_type(8)))  float  v8f;
typedef __attribute__((ext_vector_type(4)))  float  v4f;

// ---------------------------------------------------------------------------
// Fragment loaders.
// A-matrix (16x32 bf16, row-major, K contiguous): per ISA layout, lanes 0-15
// hold K = [0..7] and [16..23]; lanes 16-31 hold K = [8..15] and [24..31].
// Each half is 8 contiguous bf16 = one 16B load.
// B-matrix (32x16 bf16) fed from a [N,K] (transposed-weight) array: lane n
// holds the same K as A: [0..7],[16..23] (lanes 0-15) or [8..15],[24..31].
// ---------------------------------------------------------------------------
__device__ __forceinline__ v16bf load_a_bf16(const bf16_t* __restrict__ row,
                                             int kb, int lane) {
  int sel = (lane & 16) >> 1;            // 0 or 8
  const bf16_t* p = row + kb + sel;
  v8bf lo = *(const v8bf*)(p);
  v8bf hi = *(const v8bf*)(p + 16);
  v16bf r;
#pragma unroll
  for (int i = 0; i < 8; ++i) { r[i] = lo[i]; r[i + 8] = hi[i]; }
  return r;
}

__device__ __forceinline__ v16bf load_a_f32(const float* __restrict__ row,
                                            int kb, int lane) {
  int sel = (lane & 16) >> 1;            // 0 or 8
  const float* p = row + kb + sel;
  v4f f0 = *(const v4f*)(p);
  v4f f1 = *(const v4f*)(p + 4);
  v4f f2 = *(const v4f*)(p + 16);
  v4f f3 = *(const v4f*)(p + 20);
  v16bf r;
#pragma unroll
  for (int i = 0; i < 4; ++i) {
    r[i]      = (bf16_t)f0[i];
    r[i + 4]  = (bf16_t)f1[i];
    r[i + 8]  = (bf16_t)f2[i];
    r[i + 12] = (bf16_t)f3[i];
  }
  return r;
}

__device__ __forceinline__ v16bf load_b_bf16(const bf16_t* __restrict__ row,
                                             int kb, int lane) {
  const bf16_t* p = row + kb + ((lane & 16) >> 1);
  v8bf lo = *(const v8bf*)(p);
  v8bf hi = *(const v8bf*)(p + 16);
  v16bf r;
#pragma unroll
  for (int i = 0; i < 8; ++i) { r[i] = lo[i]; r[i + 8] = hi[i]; }
  return r;
}

__device__ __forceinline__ v8f wmma_bf16(v16bf a, v16bf b, v8f c) {
  // 8 args: (neg_a, A, neg_b, B, c_mod, C, reuse_a, reuse_b)
  return __builtin_amdgcn_wmma_f32_16x16x32_bf16(false, a, false, b,
                                                 (short)0, c, false, false);
}

// ---------------------------------------------------------------------------
// f32 -> bf16 elementwise convert (vector loads/stores)
// ---------------------------------------------------------------------------
__global__ void __launch_bounds__(256) to_bf16_kernel(const float* __restrict__ src,
                                                      bf16_t* __restrict__ dst, int n) {
  int idx = (blockIdx.x * blockDim.x + threadIdx.x) * 4;
  if (idx >= n) return;
  v4f v = *(const v4f*)(src + idx);
  v4bf o;
#pragma unroll
  for (int i = 0; i < 4; ++i) o[i] = (bf16_t)v[i];
  *(v4bf*)(dst + idx) = o;
}

// ---------------------------------------------------------------------------
// f32 [K,N] -> bf16 [N,K] transpose-convert (LDS tile, padded to avoid bank
// conflicts)
// ---------------------------------------------------------------------------
__global__ void transpose_bf16_kernel(const float* __restrict__ W,
                                      bf16_t* __restrict__ Wt, int K, int N) {
  __shared__ float tile[32][33];
  int nb = blockIdx.x * 32, kb = blockIdx.y * 32;
  int tx = threadIdx.x, ty = threadIdx.y;        // block (32,8)
  for (int i = ty; i < 32; i += 8)
    tile[i][tx] = W[(size_t)(kb + i) * N + nb + tx];
  __syncthreads();
  for (int i = ty; i < 32; i += 8)
    Wt[(size_t)(nb + i) * K + kb + tx] = (bf16_t)tile[tx][i];
}

// ---------------------------------------------------------------------------
// Generic WMMA GEMM: C[M,N] = A[M,K](bf16,row-major) * Bt[N,K](bf16) + bias
// 2x2 register blocking: each wave owns a 32x32 output block (4 accumulators),
// so each K-step issues 4 WMMAs from 2 A-fragments + 2 B-fragments.
// Block = 8 waves arranged 2(M) x 4(N) -> block tile 64 x 128.
// mode 0: store f32 row-major; mode 1: store bf16 row-major;
// mode 2: store bf16 per-head-transposed [B,H,64,S] (V projection only).
// ---------------------------------------------------------------------------
__global__ void __launch_bounds__(256) gemm_wmma(const bf16_t* __restrict__ A,
                                                 const bf16_t* __restrict__ Bt,
                                                 const float* __restrict__ bias,
                                                 float* __restrict__ Cf,
                                                 bf16_t* __restrict__ Cb,
                                                 int M, int N, int K,
                                                 int mode, int relu) {
  int lane = threadIdx.x & 31;
  int wave = threadIdx.x >> 5;
  int m0 = blockIdx.y * 64 + (wave >> 2) * 32;
  int n0 = blockIdx.x * 128 + (wave & 3) * 32;
  if (m0 >= M || n0 >= N) return;          // uniform per wave (EXEC stays all-1)

  const bf16_t* arow0 = A + (size_t)(m0 + (lane & 15)) * K;
  const bf16_t* arow1 = arow0 + (size_t)16 * K;
  const bf16_t* brow0 = Bt + (size_t)(n0 + (lane & 15)) * K;
  const bf16_t* brow1 = brow0 + (size_t)16 * K;
  v8f acc00 = {}, acc01 = {}, acc10 = {}, acc11 = {};
  for (int kb = 0; kb < K; kb += 32) {
    v16bf a0 = load_a_bf16(arow0, kb, lane);
    v16bf a1 = load_a_bf16(arow1, kb, lane);
    v16bf b0 = load_b_bf16(brow0, kb, lane);
    v16bf b1 = load_b_bf16(brow1, kb, lane);
    acc00 = wmma_bf16(a0, b0, acc00);
    acc01 = wmma_bf16(a0, b1, acc01);
    acc10 = wmma_bf16(a1, b0, acc10);
    acc11 = wmma_bf16(a1, b1, acc11);
  }

#pragma unroll
  for (int mi = 0; mi < 2; ++mi) {
#pragma unroll
    for (int ni = 0; ni < 2; ++ni) {
      v8f acc = mi ? (ni ? acc11 : acc10) : (ni ? acc01 : acc00);
      int col = n0 + ni * 16 + (lane & 15);
      float bv = bias[col];
      int rbase = m0 + mi * 16 + ((lane & 16) >> 1);   // C/D VGPR layout
#pragma unroll
      for (int i = 0; i < 8; ++i) {
        float v = acc[i] + bv;
        if (relu) v = fmaxf(v, 0.0f);
        int row = rbase + i;
        if (mode == 0) {
          Cf[(size_t)row * N + col] = v;
        } else if (mode == 1) {
          Cb[(size_t)row * N + col] = (bf16_t)v;
        } else {                           // V: [B,S,H*64] -> [B,H,64,S]
          int bb = row >> 11, s = row & 2047;
          int h = col >> 6,  d = col & 63;
          Cb[((size_t)((bb << 4) + h) * 64 + d) * 2048 + s] = (bf16_t)v;
        }
      }
    }
  }
}

// ---------------------------------------------------------------------------
// Attention scores: attn[z,q,k] = (Q_h . K_h) / 8 ;  z = b*16+h, K-dim = 64
// K-matrix rows (d contiguous) directly serve as the transposed B operand.
// 2x2 blocking; block tile 64(q) x 128(k).
// ---------------------------------------------------------------------------
__global__ void __launch_bounds__(256) attn_scores(const bf16_t* __restrict__ Q,
                                                   const bf16_t* __restrict__ Km,
                                                   float* __restrict__ attn) {
  int lane = threadIdx.x & 31;
  int wave = threadIdx.x >> 5;
  int z = blockIdx.z, b = z >> 4, h = z & 15;
  int m0 = blockIdx.y * 64 + (wave >> 2) * 32;   // query tile
  int n0 = blockIdx.x * 128 + (wave & 3) * 32;   // key tile
  const bf16_t* qrow0 = Q  + (size_t)(b * 2048 + m0 + (lane & 15)) * 1024 + h * 64;
  const bf16_t* qrow1 = qrow0 + (size_t)16 * 1024;
  const bf16_t* krow0 = Km + (size_t)(b * 2048 + n0 + (lane & 15)) * 1024 + h * 64;
  const bf16_t* krow1 = krow0 + (size_t)16 * 1024;
  v8f acc00 = {}, acc01 = {}, acc10 = {}, acc11 = {};
#pragma unroll
  for (int kb = 0; kb < 64; kb += 32) {
    v16bf a0 = load_a_bf16(qrow0, kb, lane);
    v16bf a1 = load_a_bf16(qrow1, kb, lane);
    v16bf b0 = load_b_bf16(krow0, kb, lane);
    v16bf b1 = load_b_bf16(krow1, kb, lane);
    acc00 = wmma_bf16(a0, b0, acc00);
    acc01 = wmma_bf16(a0, b1, acc01);
    acc10 = wmma_bf16(a1, b0, acc10);
    acc11 = wmma_bf16(a1, b1, acc11);
  }

  float* out = attn + (size_t)z * 2048 * 2048;
#pragma unroll
  for (int mi = 0; mi < 2; ++mi) {
#pragma unroll
    for (int ni = 0; ni < 2; ++ni) {
      v8f acc = mi ? (ni ? acc11 : acc10) : (ni ? acc01 : acc00);
      int col = n0 + ni * 16 + (lane & 15);
      int rbase = m0 + mi * 16 + ((lane & 16) >> 1);
#pragma unroll
      for (int i = 0; i < 8; ++i)
        out[(size_t)(rbase + i) * 2048 + col] = acc[i] * 0.125f;
    }
  }
}

// ---------------------------------------------------------------------------
// Row softmax over 2048 keys, in place. One 256-thread block per row.
// ---------------------------------------------------------------------------
__global__ void __launch_bounds__(256) softmax_kernel(float* __restrict__ attn) {
  __shared__ float red[256];
  size_t row = blockIdx.x;
  float* p = attn + row * 2048;
  int t = threadIdx.x;
  float v[8];
  float mx = -3.402823466e38f;
#pragma unroll
  for (int i = 0; i < 8; ++i) { v[i] = p[t + i * 256]; mx = fmaxf(mx, v[i]); }
  red[t] = mx; __syncthreads();
  for (int s = 128; s > 0; s >>= 1) { if (t < s) red[t] = fmaxf(red[t], red[t + s]); __syncthreads(); }
  mx = red[0]; __syncthreads();
  float s = 0.f;
#pragma unroll
  for (int i = 0; i < 8; ++i) { v[i] = __expf(v[i] - mx); s += v[i]; }
  red[t] = s; __syncthreads();
  for (int st = 128; st > 0; st >>= 1) { if (t < st) red[t] += red[t + st]; __syncthreads(); }
  float inv = 1.0f / red[0];
#pragma unroll
  for (int i = 0; i < 8; ++i) p[t + i * 256] = v[i] * inv;
}

// ---------------------------------------------------------------------------
// Context: ctx[b,q,h*64+d] = sum_k attn[z,q,k] * V[b,k,h*64+d]
// A = attn (f32, converted to bf16 on load); B = Vt[z][d][k] (head-transposed).
// M = 2048 (q), N = 64 (d), K = 2048 (keys). Waves arranged 4(M) x 2(N);
// block tile 128 x 64, 2x2 blocking per wave.
// ---------------------------------------------------------------------------
__global__ void __launch_bounds__(256) attn_ctx(const float* __restrict__ attn,
                                                const bf16_t* __restrict__ Vt,
                                                bf16_t* __restrict__ ctx) {
  int lane = threadIdx.x & 31;
  int wave = threadIdx.x >> 5;
  int z = blockIdx.z, b = z >> 4, h = z & 15;
  int m0 = blockIdx.y * 128 + (wave >> 1) * 32;
  int n0 = (wave & 1) * 32;                      // N = 64, gridDim.x == 1
  const float*  arow0 = attn + (size_t)z * 2048 * 2048 + (size_t)(m0 + (lane & 15)) * 2048;
  const float*  arow1 = arow0 + (size_t)16 * 2048;
  const bf16_t* brow0 = Vt   + (size_t)z * 64 * 2048   + (size_t)(n0 + (lane & 15)) * 2048;
  const bf16_t* brow1 = brow0 + (size_t)16 * 2048;
  v8f acc00 = {}, acc01 = {}, acc10 = {}, acc11 = {};
  for (int kb = 0; kb < 2048; kb += 32) {
    v16bf a0 = load_a_f32(arow0, kb, lane);
    v16bf a1 = load_a_f32(arow1, kb, lane);
    v16bf b0 = load_b_bf16(brow0, kb, lane);
    v16bf b1 = load_b_bf16(brow1, kb, lane);
    acc00 = wmma_bf16(a0, b0, acc00);
    acc01 = wmma_bf16(a0, b1, acc01);
    acc10 = wmma_bf16(a1, b0, acc10);
    acc11 = wmma_bf16(a1, b1, acc11);
  }

#pragma unroll
  for (int mi = 0; mi < 2; ++mi) {
#pragma unroll
    for (int ni = 0; ni < 2; ++ni) {
      v8f acc = mi ? (ni ? acc11 : acc10) : (ni ? acc01 : acc00);
      int d = n0 + ni * 16 + (lane & 15);
      int rbase = m0 + mi * 16 + ((lane & 16) >> 1);
#pragma unroll
      for (int i = 0; i < 8; ++i) {
        int q = rbase + i;
        ctx[(size_t)(b * 2048 + q) * 1024 + h * 64 + d] = (bf16_t)acc[i];
      }
    }
  }
}

// ---------------------------------------------------------------------------
// Fused residual add + LayerNorm over D=1024. One block per row.
// Writes f32 result and (optionally) a bf16 copy for the next GEMM.
// ---------------------------------------------------------------------------
__global__ void __launch_bounds__(256) add_ln_kernel(const float* __restrict__ X,
                                                     const float* __restrict__ Y,
                                                     const float* __restrict__ gamma,
                                                     const float* __restrict__ beta,
                                                     float* __restrict__ outF,
                                                     bf16_t* __restrict__ outB) {
  __shared__ float red[256];
  size_t row = blockIdx.x;
  const float* xp = X + row * 1024;
  const float* yp = Y + row * 1024;
  int t = threadIdx.x;
  float v[4];
  float s = 0.f;
#pragma unroll
  for (int i = 0; i < 4; ++i) { int idx = t + i * 256; v[i] = xp[idx] + yp[idx]; s += v[i]; }
  red[t] = s; __syncthreads();
  for (int st = 128; st > 0; st >>= 1) { if (t < st) red[t] += red[t + st]; __syncthreads(); }
  float mean = red[0] * (1.0f / 1024.0f);
  __syncthreads();
  float s2 = 0.f;
#pragma unroll
  for (int i = 0; i < 4; ++i) { float d = v[i] - mean; s2 += d * d; }
  red[t] = s2; __syncthreads();
  for (int st = 128; st > 0; st >>= 1) { if (t < st) red[t] += red[t + st]; __syncthreads(); }
  float inv = rsqrtf(red[0] * (1.0f / 1024.0f) + 1e-10f);
#pragma unroll
  for (int i = 0; i < 4; ++i) {
    int idx = t + i * 256;
    float o = gamma[idx] * ((v[i] - mean) * inv) + beta[idx];
    outF[row * 1024 + idx] = o;
    if (outB) outB[row * 1024 + idx] = (bf16_t)o;
  }
}

// ---------------------------------------------------------------------------
// Orchestration
// ---------------------------------------------------------------------------
extern "C" void kernel_launch(void* const* d_in, const int* in_sizes, int n_in,
                              void* d_out, int out_size, void* d_ws, size_t ws_size,
                              hipStream_t stream) {
  (void)in_sizes; (void)n_in; (void)out_size; (void)ws_size;
  const float* x   = (const float*)d_in[0];
  const float* wq  = (const float*)d_in[1];
  const float* bq  = (const float*)d_in[2];
  const float* wk  = (const float*)d_in[3];
  const float* bk  = (const float*)d_in[4];
  const float* wv  = (const float*)d_in[5];
  const float* bv  = (const float*)d_in[6];
  const float* wo  = (const float*)d_in[7];
  const float* bo  = (const float*)d_in[8];
  const float* g1  = (const float*)d_in[9];
  const float* b1  = (const float*)d_in[10];
  const float* w1  = (const float*)d_in[11];
  const float* bf1 = (const float*)d_in[12];
  const float* w2  = (const float*)d_in[13];
  const float* bf2 = (const float*)d_in[14];
  const float* g2  = (const float*)d_in[15];
  const float* b2  = (const float*)d_in[16];

  constexpr int M = 8192, D = 1024, F = 4096, S = 2048;

  char* ws = (char*)d_ws;
  size_t off = 0;
  auto take = [&](size_t bytes) -> char* {
    char* p = ws + off;
    off += (bytes + 255) & ~(size_t)255;
    return p;
  };
  bf16_t* x_bf   = (bf16_t*)take((size_t)M * D * 2);
  bf16_t* wq_t   = (bf16_t*)take((size_t)D * D * 2);
  bf16_t* wk_t   = (bf16_t*)take((size_t)D * D * 2);
  bf16_t* wv_t   = (bf16_t*)take((size_t)D * D * 2);
  bf16_t* wo_t   = (bf16_t*)take((size_t)D * D * 2);
  bf16_t* w1_t   = (bf16_t*)take((size_t)D * F * 2);
  bf16_t* w2_t   = (bf16_t*)take((size_t)F * D * 2);
  bf16_t* q_bf   = (bf16_t*)take((size_t)M * D * 2);
  bf16_t* k_bf   = (bf16_t*)take((size_t)M * D * 2);
  bf16_t* vt_bf  = (bf16_t*)take((size_t)M * D * 2);   // [B,H,64,S]
  bf16_t* ctx_bf = (bf16_t*)take((size_t)M * D * 2);
  float*  proj   = (float*)take((size_t)M * D * 4);
  float*  out1f  = (float*)take((size_t)M * D * 4);
  bf16_t* out1b  = (bf16_t*)take((size_t)M * D * 2);
  bf16_t* hid_bf = q_bf;     // reuse 64MB q/k/vt/ctx block after attention
  float*  ff     = proj;     // reuse O-proj f32 buffer for FFN2 output

  float* out2 = (float*)d_out;                 // [4,2048,1024]
  float* attn = (float*)d_out + (size_t)M * D; // [4,16,2048,2048]

  // --- one-time conversions: bf16 activations, transposed bf16 weights ---
  to_bf16_kernel<<<(M * D / 4 + 255) / 256, 256, 0, stream>>>(x, x_bf, M * D);
  dim3 tb(32, 8);
  transpose_bf16_kernel<<<dim3(D / 32, D / 32), tb, 0, stream>>>(wq, wq_t, D, D);
  transpose_bf16_kernel<<<dim3(D / 32, D / 32), tb, 0, stream>>>(wk, wk_t, D, D);
  transpose_bf16_kernel<<<dim3(D / 32, D / 32), tb, 0, stream>>>(wv, wv_t, D, D);
  transpose_bf16_kernel<<<dim3(D / 32, D / 32), tb, 0, stream>>>(wo, wo_t, D, D);
  transpose_bf16_kernel<<<dim3(F / 32, D / 32), tb, 0, stream>>>(w1, w1_t, D, F);
  transpose_bf16_kernel<<<dim3(D / 32, F / 32), tb, 0, stream>>>(w2, w2_t, F, D);

  // --- QKV projections (WMMA) ---
  dim3 gDD(D / 128, M / 64);
  gemm_wmma<<<gDD, 256, 0, stream>>>(x_bf, wq_t, bq, nullptr, q_bf,  M, D, D, 1, 0);
  gemm_wmma<<<gDD, 256, 0, stream>>>(x_bf, wk_t, bk, nullptr, k_bf,  M, D, D, 1, 0);
  gemm_wmma<<<gDD, 256, 0, stream>>>(x_bf, wv_t, bv, nullptr, vt_bf, M, D, D, 2, 0);

  // --- attention ---
  attn_scores<<<dim3(S / 128, S / 64, 64), 256, 0, stream>>>(q_bf, k_bf, attn);
  softmax_kernel<<<64 * S, 256, 0, stream>>>(attn);
  attn_ctx<<<dim3(1, S / 128, 64), 256, 0, stream>>>(attn, vt_bf, ctx_bf);

  // --- output projection + residual LayerNorm ---
  gemm_wmma<<<gDD, 256, 0, stream>>>(ctx_bf, wo_t, bo, proj, nullptr, M, D, D, 0, 0);
  add_ln_kernel<<<M, 256, 0, stream>>>(x, proj, g1, b1, out1f, out1b);

  // --- FFN ---
  gemm_wmma<<<dim3(F / 128, M / 64), 256, 0, stream>>>(out1b, w1_t, bf1, nullptr, hid_bf, M, F, D, 1, 1);
  gemm_wmma<<<gDD, 256, 0, stream>>>(hid_bf, w2_t, bf2, ff, nullptr, M, D, F, 0, 0);
  add_ln_kernel<<<M, 256, 0, stream>>>(out1f, ff, g2, b2, out2, nullptr);
}
